// DKVMN_57208964383531
// MI455X (gfx1250) — matrix operand order repaired and checked by reading, not compiled, not run on hardware
//
#include <hip/hip_runtime.h>
#include <hip/hip_bf16.h>
#include <math.h>

// ---------------------------------------------------------------------------
// DKVMN for MI455X (gfx1250, wave32, WMMA).
//   GEMM phase: bf16 V_WMMA_F32_16X16X32_BF16, f32 accumulate.
//   Scan phase: bandwidth-bound streaming writer (327.7 MB output).
// Workspace layout (requires ~16.6 MB of d_ws):
//   [0)        MkT  bf16 64x128   (Mk transposed, N padded 50->64)
//   [16384)    MvT  bf16 128x64   (Mv transposed, K padded 50->64)
//   [32768)    fWb  bf16 128x256  (f_W, already N-major/K-contig)
//   [98304)    eWb  bf16 128x128
//   [131072)   aWb  bf16 128x128
//   [163840)   wt   f32  12800x64
//   [3440640)  et   f32  12800x128
//   [9994240)  at   f32  12800x128   (ends at 16547840 bytes)
// ---------------------------------------------------------------------------

typedef __attribute__((ext_vector_type(16))) __bf16 v16bf;
typedef __attribute__((ext_vector_type(8)))  __bf16 v8bf;
typedef __attribute__((ext_vector_type(8)))  float  v8f;

#define B_DIM  64
#define T_DIM  200
#define NQ     1000
#define DK     128
#define DV     128
#define NS     50
#define NSP    64
#define NROWS  (B_DIM * T_DIM)   // 12800

// A fragment (16xK tile, row-major bf16 in LDS, stride in elements).
// Lane layout: M = lane%16, half = lane/16.
//   elems 0..7  -> K = k0 + half*8 + i
//   elems 8..15 -> K = k0 + 16 + half*8 + i
__device__ __forceinline__ v16bf load_a_lds(const __bf16* smem, int stride,
                                            int k0, int lane) {
  int m = lane & 15, half = lane >> 4;
  const v8bf* p = (const v8bf*)(smem + m * stride + k0 + half * 8);
  v8bf lo = p[0];
  v8bf hi = p[2];   // +16 elements
  v16bf a;
#pragma unroll
  for (int i = 0; i < 8; ++i) { a[i] = lo[i]; a[i + 8] = hi[i]; }
  return a;
}

// B fragment from global weights stored transposed: BT[n][k], k contiguous.
// Lane layout: N = lane%16, elems 0..7 -> K = k0 + half*8 + i, elems 8..15 -> K = k0 + 16 + half*8 + i.
__device__ __forceinline__ v16bf load_b_glb(const __bf16* BT, int ldk,
                                            int n0, int k0, int lane) {
  int n = lane & 15, half = lane >> 4;
  const v8bf* p = (const v8bf*)(BT + (size_t)(n0 + n) * ldk + k0 + half * 8);
  v8bf lo = p[0];
  v8bf hi = p[2];
  v16bf b;
#pragma unroll
  for (int i = 0; i < 8; ++i) { b[i] = lo[i]; b[i + 8] = hi[i]; }
  return b;
}

__device__ __forceinline__ v8f wmma_bf16(v16bf a, v16bf b, v8f c) {
  return __builtin_amdgcn_wmma_f32_16x16x32_bf16(
      /*neg_a=*/false, a, /*neg_b=*/false, b,
      /*c_mod=*/(short)0, c, /*reuse_a=*/false, /*reuse_b=*/false);
}

__device__ __forceinline__ float sigm(float x) {
  return 1.f / (1.f + __expf(-x));
}

// ---------------------------------------------------------------------------
// Kernel 0: weight conversion / transposition to bf16.
// ---------------------------------------------------------------------------
__global__ __launch_bounds__(256) void dkvmn_prep(
    const float* __restrict__ Mk, const float* __restrict__ Mv,
    const float* __restrict__ fW, const float* __restrict__ eW,
    const float* __restrict__ aW,
    __bf16* __restrict__ MkT, __bf16* __restrict__ MvT,
    __bf16* __restrict__ fWb, __bf16* __restrict__ eWb,
    __bf16* __restrict__ aWb) {
  int i = blockIdx.x * 256 + threadIdx.x;          // 0 .. 32767
  if (i < NSP * DK) {                              // MkT[n][k] = Mk[k][n]
    int n = i >> 7, k = i & 127;
    MkT[i] = (n < NS) ? (__bf16)Mk[k * NS + n] : (__bf16)0.f;
  }
  if (i < DV * NSP) {                              // MvT[n][k] = Mv[k][n]
    int n = i >> 6, k = i & 63;
    MvT[i] = (k < NS) ? (__bf16)Mv[k * DV + n] : (__bf16)0.f;
  }
  if (i < 128 * 256) fWb[i] = (__bf16)fW[i];       // f_W already (N,K)
  if (i < 128 * 128) {
    eWb[i] = (__bf16)eW[i];
    aWb[i] = (__bf16)aW[i];
  }
}

// ---------------------------------------------------------------------------
// Kernel 1: fused embed-gather + 4 chained WMMA GEMMs per 16-row tile.
// One wave (32 threads) per block; 800 blocks.
// ---------------------------------------------------------------------------
__global__ __launch_bounds__(32) void dkvmn_fused_gemm(
    const int* __restrict__ q, const int* __restrict__ r,
    const float* __restrict__ k_emb, const float* __restrict__ v_emb,
    const __bf16* __restrict__ MkT, const __bf16* __restrict__ MvT,
    const __bf16* __restrict__ fWb, const __bf16* __restrict__ eWb,
    const __bf16* __restrict__ aWb,
    const float* __restrict__ f_b, const float* __restrict__ p_W,
    const float* __restrict__ p_b, const float* __restrict__ e_b,
    const float* __restrict__ a_b,
    float* __restrict__ wt_out, float* __restrict__ et_out,
    float* __restrict__ at_out, float* __restrict__ p_out) {
  __shared__ __align__(16) __bf16 sK[16 * DK];    // kt tile (bf16)
  __shared__ __align__(16) __bf16 sV[16 * DV];    // vt tile (bf16)
  __shared__ __align__(16) __bf16 sF[16 * 256];   // f_in = [rt | kt]
  __shared__ __align__(16) __bf16 sW[16 * NSP];   // softmax weights, padded
  __shared__ __align__(16) float  sS[16 * 128];   // f32 scratch: logits, ft
  __shared__ int sQ[16], sR[16];

  const int lane = threadIdx.x;
  const int row0 = blockIdx.x * 16;

  if (lane < 16) {
    sQ[lane] = q[row0 + lane];
    sR[lane] = r[row0 + lane];
  }
  __syncthreads();

  // ---- gather kt / vt, convert to bf16; kt also feeds f_in[:,128:256] ----
  for (int i = lane; i < 16 * DK; i += 32) {
    int m = i >> 7, c = i & 127;
    int qv = sQ[m], rv = sR[m];
    __bf16 kb = (__bf16)k_emb[qv * DK + c];
    sK[m * DK + c] = kb;
    sF[m * 256 + 128 + c] = kb;
    sV[m * DV + c] = (__bf16)v_emb[(qv + NQ * rv) * DV + c];
  }
  __syncthreads();

  // ---- logits = kt @ Mk : 16x64 (padded), 4 N-tiles x 4 K-steps ----
  for (int nt = 0; nt < 4; ++nt) {
    v8f acc = {};
#pragma unroll
    for (int kk = 0; kk < 4; ++kk) {
      v16bf a = load_a_lds(sK, DK, kk * 32, lane);
      v16bf b = load_b_glb(MkT, DK, nt * 16, kk * 32, lane);
      acc = wmma_bf16(a, b, acc);
    }
    int n = nt * 16 + (lane & 15), half = lane >> 4;
#pragma unroll
    for (int rr = 0; rr < 8; ++rr) sS[(half * 8 + rr) * 64 + n] = acc[rr];
  }
  __syncthreads();

  // ---- row softmax over the 50 real slots (one lane per row) ----
  if (lane < 16) {
    float mx = -1e30f;
    for (int n = 0; n < NS; ++n) mx = fmaxf(mx, sS[lane * 64 + n]);
    float sum = 0.f;
    for (int n = 0; n < NS; ++n) {
      float e = __expf(sS[lane * 64 + n] - mx);
      sS[lane * 64 + n] = e;
      sum += e;
    }
    float inv = 1.f / sum;
    int gr = row0 + lane;
    for (int n = 0; n < NSP; ++n) {
      float w = (n < NS) ? sS[lane * 64 + n] * inv : 0.f;
      sW[lane * NSP + n] = (__bf16)w;
      wt_out[(size_t)gr * NSP + n] = w;
    }
  }
  __syncthreads();

  // ---- rt = wt @ Mv : 16x128 -> f_in[:,0:128], 8 N-tiles x 2 K-steps ----
  for (int nt = 0; nt < 8; ++nt) {
    v8f acc = {};
#pragma unroll
    for (int kk = 0; kk < 2; ++kk) {
      v16bf a = load_a_lds(sW, NSP, kk * 32, lane);
      v16bf b = load_b_glb(MvT, NSP, nt * 16, kk * 32, lane);
      acc = wmma_bf16(a, b, acc);
    }
    int n = nt * 16 + (lane & 15), half = lane >> 4;
#pragma unroll
    for (int rr = 0; rr < 8; ++rr)
      sF[(half * 8 + rr) * 256 + n] = (__bf16)acc[rr];
  }
  __syncthreads();

  // ---- ft = tanh(f_in @ f_W^T + f_b) : 16x128, 8 N-tiles x 8 K-steps ----
  for (int nt = 0; nt < 8; ++nt) {
    v8f acc = {};
#pragma unroll
    for (int kk = 0; kk < 8; ++kk) {
      v16bf a = load_a_lds(sF, 256, kk * 32, lane);
      v16bf b = load_b_glb(fWb, 256, nt * 16, kk * 32, lane);
      acc = wmma_bf16(a, b, acc);
    }
    int n = nt * 16 + (lane & 15), half = lane >> 4;
    float bias = f_b[n];
#pragma unroll
    for (int rr = 0; rr < 8; ++rr)
      sS[(half * 8 + rr) * 128 + n] = tanhf(acc[rr] + bias);
  }
  __syncthreads();

  // ---- p = sigmoid(ft . p_W + p_b), one lane per row ----
  if (lane < 16) {
    float d = p_b[0];
    for (int c = 0; c < 128; ++c) d += sS[lane * 128 + c] * p_W[c];
    p_out[row0 + lane] = sigm(d);
  }

  // ---- et = sigmoid(vt @ e_W^T + e_b), at = tanh(vt @ a_W^T + a_b) ----
  for (int nt = 0; nt < 8; ++nt) {
    v8f accE = {}, accA = {};
#pragma unroll
    for (int kk = 0; kk < 4; ++kk) {
      v16bf a  = load_a_lds(sV, DV, kk * 32, lane);
      v16bf be = load_b_glb(eWb, DK, nt * 16, kk * 32, lane);
      v16bf ba = load_b_glb(aWb, DK, nt * 16, kk * 32, lane);
      accE = wmma_bf16(a, be, accE);
      accA = wmma_bf16(a, ba, accA);
    }
    int n = nt * 16 + (lane & 15), half = lane >> 4;
    float eb = e_b[n], ab = a_b[n];
#pragma unroll
    for (int rr = 0; rr < 8; ++rr) {
      int gr = row0 + half * 8 + rr;
      et_out[(size_t)gr * 128 + n] = sigm(accE[rr] + eb);
      at_out[(size_t)gr * 128 + n] = tanhf(accA[rr] + ab);
    }
  }
}

// ---------------------------------------------------------------------------
// Kernel 2: sequential memory-state scan. One block per batch element.
// 256 threads x 25 state elements (50x128) in registers; coalesced stores.
// Writes 327.7 MB total -> HBM-bound (~14 us floor at 23.3 TB/s).
// ---------------------------------------------------------------------------
__global__ __launch_bounds__(256) void dkvmn_scan(
    const float* __restrict__ wt, const float* __restrict__ et,
    const float* __restrict__ at, const float* __restrict__ Mv,
    float* __restrict__ MvS) {
  __shared__ float sw[NSP], se[DV], sa[DV];
  const int b = blockIdx.x, tid = threadIdx.x;

  float m[25];
#pragma unroll
  for (int j = 0; j < 25; ++j) m[j] = Mv[tid + j * 256];

  float* out = MvS + (size_t)b * T_DIM * (NS * DV);
  for (int t = 0; t < T_DIM; ++t) {
    const int row = b * T_DIM + t;
    __syncthreads();
    if (tid < 128) {
      se[tid] = et[(size_t)row * 128 + tid];
      if (tid < NS) sw[tid] = wt[(size_t)row * NSP + tid];
    } else {
      sa[tid - 128] = at[(size_t)row * 128 + (tid - 128)];
    }
    __syncthreads();
#pragma unroll
    for (int j = 0; j < 25; ++j) {
      int idx = tid + j * 256;
      int n = idx >> 7, d = idx & 127;
      float w = sw[n];
      m[j] = m[j] * (1.f - w * se[d]) + w * sa[d];
      out[(size_t)t * (NS * DV) + idx] = m[j];
    }
  }
}

// ---------------------------------------------------------------------------
extern "C" void kernel_launch(void* const* d_in, const int* in_sizes, int n_in,
                              void* d_out, int out_size, void* d_ws,
                              size_t ws_size, hipStream_t stream) {
  const int*   q     = (const int*)d_in[0];
  const int*   r     = (const int*)d_in[1];
  const float* k_emb = (const float*)d_in[2];
  const float* v_emb = (const float*)d_in[3];
  const float* Mk    = (const float*)d_in[4];
  const float* Mv    = (const float*)d_in[5];
  const float* f_W   = (const float*)d_in[6];
  const float* f_b   = (const float*)d_in[7];
  const float* p_W   = (const float*)d_in[8];
  const float* p_b   = (const float*)d_in[9];
  const float* e_W   = (const float*)d_in[10];
  const float* e_b   = (const float*)d_in[11];
  const float* a_W   = (const float*)d_in[12];
  const float* a_b   = (const float*)d_in[13];

  char* ws = (char*)d_ws;
  __bf16* MkT = (__bf16*)(ws + 0);
  __bf16* MvT = (__bf16*)(ws + 16384);
  __bf16* fWb = (__bf16*)(ws + 32768);
  __bf16* eWb = (__bf16*)(ws + 98304);
  __bf16* aWb = (__bf16*)(ws + 131072);
  float*  wt  = (float*)(ws + 163840);
  float*  et  = (float*)(ws + 163840 + 3276800);
  float*  at  = (float*)(ws + 163840 + 3276800 + 6553600);

  float* p_out = (float*)d_out;           // (B,T) = 12800 floats
  float* MvS   = (float*)d_out + NROWS;   // (B,T,50,128)

  dkvmn_prep<<<128, 256, 0, stream>>>(Mk, Mv, f_W, e_W, a_W,
                                      MkT, MvT, fWb, eWb, aWb);
  dkvmn_fused_gemm<<<NROWS / 16, 32, 0, stream>>>(
      q, r, k_emb, v_emb, MkT, MvT, fWb, eWb, aWb,
      f_b, p_W, p_b, e_b, a_b, wt, et, at, p_out);
  dkvmn_scan<<<B_DIM, 256, 0, stream>>>(wt, et, at, Mv, MvS);
}
